// FeedbackTransformerBlock_32727650795905
// MI455X (gfx1250) — matrix operand order repaired and checked by reading, not compiled, not run on hardware
//
#include <hip/hip_runtime.h>
#include <hip/hip_bf16.h>
#include <math.h>

typedef __attribute__((ext_vector_type(16))) __bf16 v16bf;
typedef __attribute__((ext_vector_type(8)))  float  v8f;

#define D_MODEL 1024
#define N_HEADS 16
#define HEAD_DIM 64
#define D_FF 4096
#define BATCH 128
#define MEM_LEN 1024
#define LN_EPS 1e-5f

// ---------------------------------------------------------------------------
// CDNA5 async global->LDS copy (ASYNCcnt-tracked, no VGPR round trip).
// GV mode: 64-bit global address in a VGPR pair, LDS byte offset in a VGPR.
// ---------------------------------------------------------------------------
__device__ __forceinline__ void async_ld_b128(unsigned lds_off, const void* gptr) {
  asm volatile("global_load_async_to_lds_b128 %0, %1, off"
               :: "v"(lds_off), "v"(gptr) : "memory");
}
__device__ __forceinline__ void wait_async(bool more_in_flight) {
  if (more_in_flight) asm volatile("s_wait_asynccnt 16" ::: "memory");
  else                asm volatile("s_wait_asynccnt 0"  ::: "memory");
}

// ---------------------------------------------------------------------------
// GEMM: C[M,N] = A[M,K] @ B[K,N] + bias (optional ReLU)
// One wave32 per 16x64 output strip: 4 fp32 accumulators. A fragment gathered
// once per K-step; all four B fragments are materialized into distinct live
// registers BEFORE the four back-to-back v_wmma_f32_16x16x32_bf16, so the
// WMMAs share A (read-read), write disjoint D, read disjoint B -> no
// WMMA->VALU WAR hazard NOPs between them.
// B tile (32x64 fp32) staged via GLOBAL_LOAD_ASYNC_TO_LDS_B128, double
// buffered and pipelined with s_wait_asynccnt. A tile staged via regular
// loads into a padded (stride 33) LDS buffer for a conflict-free gather.
// ---------------------------------------------------------------------------
__global__ __launch_bounds__(32) void gemm_wmma(
    const float* __restrict__ A, const float* __restrict__ B,
    const float* __restrict__ bias, float* __restrict__ C,
    int M, int N, int K, int relu)
{
  __shared__ float As[16 * 33];        // 16 x 32 fp32, padded
  __shared__ float Bs[2][32 * 64];     // double-buffered 32 x 64 fp32

  const int t      = threadIdx.x;      // 0..31
  const int tilesN = N >> 6;           // N / 64
  const int tm     = blockIdx.x / tilesN;
  const int tn     = blockIdx.x % tilesN;
  const int half   = t >> 4;           // 0 or 1
  const int lane16 = t & 15;

  v8f acc0 = {}, acc1 = {}, acc2 = {}, acc3 = {};

  // --- issue async B-tile copy for one K-step into buffer `buf` -----------
  auto issue_B = [&](int k0, int buf) {
    unsigned base = (unsigned)(size_t)(&Bs[buf][0]);
    #pragma unroll
    for (int i = 0; i < 16; ++i) {
      int lin = i * 32 + t;            // 0..511 float4 slots of the 32x64 tile
      int r   = lin >> 4;              // row 0..31 (k)
      int c4  = lin & 15;              // float4 col 0..15 (n/4)
      const void* g = (const void*)(B + (size_t)(k0 + r) * N + tn * 64 + c4 * 4);
      async_ld_b128(base + (unsigned)(r * 64 + c4 * 4) * 4u, g);
    }
  };

  issue_B(0, 0);

  for (int k0 = 0; k0 < K; k0 += 32) {
    const int  buf  = (k0 >> 5) & 1;
    const bool more = (k0 + 32) < K;
    if (more) issue_B(k0 + 32, buf ^ 1);   // prefetch next tile while we work

    // --- A tile (16 x 32 fp32): regular float4 loads, padded LDS ---------
    const float4* Ag = (const float4*)(A + (size_t)(tm * 16) * K + k0);
    #pragma unroll
    for (int i = 0; i < 4; ++i) {
      int lin = i * 32 + t;            // 0..127 float4 slots
      int r   = lin >> 3;              // row 0..15
      int c4  = lin & 7;               // float4 col 0..7
      float4 v = Ag[(size_t)r * (K >> 2) + c4];
      float* dst = &As[r * 33 + c4 * 4];
      dst[0] = v.x; dst[1] = v.y; dst[2] = v.z; dst[3] = v.w;
    }
    __syncthreads();
    wait_async(more);                  // current B buffer is now resident

    // --- fragments per ISA 7.12.2 wave32 layouts, cvt f32->bf16 ----------
    v16bf af;
    #pragma unroll
    for (int e = 0; e < 16; ++e) {
      int ka = ((e >> 3) << 4) + (half << 3) + (e & 7);
      af[e] = (__bf16)As[lane16 * 33 + ka];
    }
    // build ALL four B fragments first (distinct live registers)
    v16bf bf0, bf1, bf2, bf3;
    #pragma unroll
    for (int e = 0; e < 16; ++e) {
      int kb = ((e >> 3) << 4) + (half << 3) + (e & 7);
      const float* brow = &Bs[buf][kb * 64 + lane16];
      bf0[e] = (__bf16)brow[0];
      bf1[e] = (__bf16)brow[16];
      bf2[e] = (__bf16)brow[32];
      bf3[e] = (__bf16)brow[48];
    }
    // four back-to-back WMMAs, shared A fragment
    acc0 = __builtin_amdgcn_wmma_f32_16x16x32_bf16(false, af, false, bf0,
                                                   (short)0, acc0, false, false);
    acc1 = __builtin_amdgcn_wmma_f32_16x16x32_bf16(false, af, false, bf1,
                                                   (short)0, acc1, false, false);
    acc2 = __builtin_amdgcn_wmma_f32_16x16x32_bf16(false, af, false, bf2,
                                                   (short)0, acc2, false, false);
    acc3 = __builtin_amdgcn_wmma_f32_16x16x32_bf16(false, af, false, bf3,
                                                   (short)0, acc3, false, false);
    __syncthreads();
  }

  // --- store D: lane holds col n=lane16; VGPR r -> row r + 8*half ---------
  #pragma unroll
  for (int j = 0; j < 4; ++j) {
    const v8f& accr = (j == 0) ? acc0 : (j == 1) ? acc1 : (j == 2) ? acc2 : acc3;
    int nn = tn * 64 + j * 16 + lane16;
    float bv = bias[nn];
    #pragma unroll
    for (int r = 0; r < 8; ++r) {
      int mm = tm * 16 + r + half * 8;
      float v = accr[r] + bv;
      if (relu) v = fmaxf(v, 0.f);
      C[(size_t)mm * N + nn] = v;
    }
  }
}

// ---------------------------------------------------------------------------
// Attention: one wave32 per (batch, head). Two memory rows per iteration:
// lanes 0-15 handle row m (float4/lane = b128 loads), lanes 16-31 row m+1.
// Half-wave online softmax states merged via shfl_xor(16) at the end.
// global_prefetch issued ~16 rows (64KB) ahead on both K and V streams.
// ---------------------------------------------------------------------------
__global__ __launch_bounds__(32) void attn_kernel(
    const float* __restrict__ q, const float* __restrict__ Kmem,
    const float* __restrict__ Vmem, const float* __restrict__ alibi,
    float* __restrict__ out)
{
  const int bh = blockIdx.x;
  const int b  = bh >> 4;
  const int h  = bh & 15;
  const int t  = threadIdx.x;
  const int hl = t & 15;         // lane within half-wave
  const int ph = t >> 4;         // row parity this half-wave handles
  const float scale = 0.125f;    // 1/sqrt(64)

  float4 qv = *(const float4*)(q + (size_t)b * D_MODEL + h * HEAD_DIM + hl * 4);
  const float* Kb = Kmem + (size_t)b * MEM_LEN * D_MODEL + h * HEAD_DIM + hl * 4;
  const float* Vb = Vmem + (size_t)b * MEM_LEN * D_MODEL + h * HEAD_DIM + hl * 4;
  const float* ab = alibi + h * MEM_LEN;

  float runmax = -INFINITY;
  float runsum = 0.f;
  float4 acc = make_float4(0.f, 0.f, 0.f, 0.f);

  for (int m2 = 0; m2 < MEM_LEN; m2 += 2) {
    const int m = m2 + ph;
    const float* kr = Kb + (size_t)m * D_MODEL;
    const float* vr = Vb + (size_t)m * D_MODEL;
    __builtin_prefetch(kr + 16 * D_MODEL, 0, 0);   // -> global_prefetch_b8
    __builtin_prefetch(vr + 16 * D_MODEL, 0, 0);

    float4 kv = *(const float4*)kr;
    float p = qv.x * kv.x + qv.y * kv.y + qv.z * kv.z + qv.w * kv.w;
    p += __shfl_xor(p, 1, 32);
    p += __shfl_xor(p, 2, 32);
    p += __shfl_xor(p, 4, 32);
    p += __shfl_xor(p, 8, 32);          // half-wave now holds its row's dot

    float s  = p * scale + ab[m];
    float nm = fmaxf(runmax, s);
    float corr = __expf(runmax - nm);   // 0 on first iteration
    float w    = __expf(s - nm);
    runsum = runsum * corr + w;
    float4 vv = *(const float4*)vr;
    acc.x = acc.x * corr + w * vv.x;
    acc.y = acc.y * corr + w * vv.y;
    acc.z = acc.z * corr + w * vv.z;
    acc.w = acc.w * corr + w * vv.w;
    runmax = nm;
  }

  // merge the two half-wave softmax states (symmetric -> both halves agree)
  float om = __shfl_xor(runmax, 16, 32);
  float os = __shfl_xor(runsum, 16, 32);
  float4 oa;
  oa.x = __shfl_xor(acc.x, 16, 32);
  oa.y = __shfl_xor(acc.y, 16, 32);
  oa.z = __shfl_xor(acc.z, 16, 32);
  oa.w = __shfl_xor(acc.w, 16, 32);

  float gm = fmaxf(runmax, om);
  float c0 = __expf(runmax - gm);
  float c1 = __expf(om - gm);
  float inv = 1.f / (runsum * c0 + os * c1);

  if (ph == 0) {
    float4 o;
    o.x = (acc.x * c0 + oa.x * c1) * inv;
    o.y = (acc.y * c0 + oa.y * c1) * inv;
    o.z = (acc.z * c0 + oa.z * c1) * inv;
    o.w = (acc.w * c0 + oa.w * c1) * inv;
    *(float4*)(out + (size_t)b * D_MODEL + h * HEAD_DIM + hl * 4) = o;
  }
}

// ---------------------------------------------------------------------------
// out[row] = LayerNorm(x[row] + y[row]) * g + be.  One 256-thread block/row.
// ---------------------------------------------------------------------------
__global__ __launch_bounds__(256) void add_layernorm(
    const float* __restrict__ x, const float* __restrict__ y,
    const float* __restrict__ g, const float* __restrict__ be,
    float* __restrict__ out)
{
  const int row = blockIdx.x;
  const int t   = threadIdx.x;

  float4 xv = ((const float4*)(x + (size_t)row * D_MODEL))[t];
  float4 yv = ((const float4*)(y + (size_t)row * D_MODEL))[t];
  float z0 = xv.x + yv.x, z1 = xv.y + yv.y, z2 = xv.z + yv.z, z3 = xv.w + yv.w;

  float s  = z0 + z1 + z2 + z3;
  float s2 = z0*z0 + z1*z1 + z2*z2 + z3*z3;
  #pragma unroll
  for (int off = 16; off; off >>= 1) {
    s  += __shfl_xor(s,  off, 32);
    s2 += __shfl_xor(s2, off, 32);
  }

  __shared__ float rs[8], rs2[8];
  int wid = t >> 5, ln = t & 31;
  if (ln == 0) { rs[wid] = s; rs2[wid] = s2; }
  __syncthreads();

  float ts = 0.f, ts2 = 0.f;
  #pragma unroll
  for (int i = 0; i < 8; ++i) { ts += rs[i]; ts2 += rs2[i]; }

  float mu  = ts  * (1.f / D_MODEL);
  float var = ts2 * (1.f / D_MODEL) - mu * mu;
  float rsd = rsqrtf(var + LN_EPS);

  float4 gv = ((const float4*)g)[t];
  float4 bv = ((const float4*)be)[t];
  float4 o;
  o.x = (z0 - mu) * rsd * gv.x + bv.x;
  o.y = (z1 - mu) * rsd * gv.y + bv.y;
  o.z = (z2 - mu) * rsd * gv.z + bv.z;
  o.w = (z3 - mu) * rsd * gv.w + bv.w;
  ((float4*)(out + (size_t)row * D_MODEL))[t] = o;
}

// ---------------------------------------------------------------------------
extern "C" void kernel_launch(void* const* d_in, const int* in_sizes, int n_in,
                              void* d_out, int out_size, void* d_ws, size_t ws_size,
                              hipStream_t stream) {
  const float* x   = (const float*)d_in[0];
  const float* mk  = (const float*)d_in[1];
  const float* mv  = (const float*)d_in[2];
  const float* al  = (const float*)d_in[3];
  const float* Wq  = (const float*)d_in[4];
  const float* bq  = (const float*)d_in[5];
  const float* Wo  = (const float*)d_in[6];
  const float* bo  = (const float*)d_in[7];
  const float* W1  = (const float*)d_in[8];
  const float* b1  = (const float*)d_in[9];
  const float* W2  = (const float*)d_in[10];
  const float* b2  = (const float*)d_in[11];
  const float* g1  = (const float*)d_in[12];
  const float* be1 = (const float*)d_in[13];
  const float* g2  = (const float*)d_in[14];
  const float* be2 = (const float*)d_in[15];
  float* out = (float*)d_out;

  float* ws   = (float*)d_ws;
  float* q    = ws;                         // 128*1024
  float* attn = q    + BATCH * D_MODEL;     // 128*1024
  float* proj = attn + BATCH * D_MODEL;     // 128*1024
  float* h    = proj + BATCH * D_MODEL;     // 128*1024
  float* ff1  = h    + BATCH * D_MODEL;     // 128*4096
  float* ff2  = ff1  + BATCH * D_FF;        // 128*1024

  dim3 w32(32);

  // q = x @ Wq + bq
  gemm_wmma<<<dim3((BATCH / 16) * (D_MODEL / 64)), w32, 0, stream>>>(
      x, Wq, bq, q, BATCH, D_MODEL, D_MODEL, 0);

  // attention (ALiBi + softmax + V aggregation), fused single pass
  attn_kernel<<<dim3(BATCH * N_HEADS), w32, 0, stream>>>(q, mk, mv, al, attn);

  // proj = attn @ Wo + bo
  gemm_wmma<<<dim3((BATCH / 16) * (D_MODEL / 64)), w32, 0, stream>>>(
      attn, Wo, bo, proj, BATCH, D_MODEL, D_MODEL, 0);

  // h = LN(x + proj)
  add_layernorm<<<dim3(BATCH), dim3(256), 0, stream>>>(x, proj, g1, be1, h);

  // ff1 = relu(h @ W1 + b1)
  gemm_wmma<<<dim3((BATCH / 16) * (D_FF / 64)), w32, 0, stream>>>(
      h, W1, b1, ff1, BATCH, D_FF, D_MODEL, 1);

  // ff2 = ff1 @ W2 + b2
  gemm_wmma<<<dim3((BATCH / 16) * (D_MODEL / 64)), w32, 0, stream>>>(
      ff1, W2, b2, ff2, BATCH, D_MODEL, D_FF, 0);

  // out = LN(h + ff2)
  add_layernorm<<<dim3(BATCH), dim3(256), 0, stream>>>(h, ff2, g2, be2, out);
}
